// MultiheadRCDA_43379169690114
// MI455X (gfx1250) — matrix operand order repaired and checked by reading, not compiled, not run on hardware
//
#include <hip/hip_runtime.h>
#include <hip/hip_bf16.h>

typedef __attribute__((ext_vector_type(16))) __bf16 v16bf;
typedef __attribute__((ext_vector_type(8)))  __bf16 v8bf;
typedef __attribute__((ext_vector_type(8)))  float  v8f;
typedef __attribute__((ext_vector_type(4)))  float  v4f;

__device__ inline v16bf load2x8(const __bf16* p0, const __bf16* p1) {
  v16bf r;
  *(uint4*)&r       = *(const uint4*)p0;
  *(((uint4*)&r)+1) = *(const uint4*)p1;
  return r;
}

__device__ inline v8f wmma_bf16(v16bf a, v16bf b, v8f c) {
  return __builtin_amdgcn_wmma_f32_16x16x32_bf16(false, a, false, b, (short)0, c,
                                                 false, false);
}

// ---------------------------------------------------------------------------
// Kernel 1: key means.  km_row[n,w,c] = mean_h key_row ; km_col[n,h,c] = mean_w key_col
// ---------------------------------------------------------------------------
__global__ __launch_bounds__(256) void means_kernel(
    const float* __restrict__ key_row, const float* __restrict__ key_col,
    float* __restrict__ km_row, float* __restrict__ km_col) {
  int idx = blockIdx.x * 256 + threadIdx.x;           // over 8*96*256
  if (idx >= 8 * 96 * 256) return;
  int c = idx & 255;
  int x = (idx >> 8) % 96;
  int n = idx / (96 * 256);
  const float* p = key_row + (size_t)n * 96 * 96 * 256 + (size_t)x * 256 + c;
  float s = 0.f;
  #pragma unroll 4
  for (int h = 0; h < 96; ++h) s += p[(size_t)h * 96 * 256];
  km_row[idx] = s * (1.f / 96.f);
  const float* q = key_col + (size_t)n * 96 * 96 * 256 + (size_t)x * 96 * 256 + c;
  float s2 = 0.f;
  #pragma unroll 4
  for (int w = 0; w < 96; ++w) s2 += q[(size_t)w * 256];
  km_col[idx] = s2 * (1.f / 96.f);
}

// ---------------------------------------------------------------------------
// Kernel 2: projection GEMM  C[M,256] = (A[M,256] @ W[256,256]^T + bias)*scale
// 64x64 tile / WG, 8 waves x 2 WMMA tiles, double-buffered LDS, vectorized
// staging (one b128 global load pair + one b128 LDS store per thread per step).
// mode: 0 -> f32 to Cf ; 1 -> bf16 to Cb (row-major) ;
//       2 -> bf16 to Cb in VT layout [n][h][e][d][w] (for the value proj).
// ---------------------------------------------------------------------------
__global__ __launch_bounds__(256) void gemm_proj_kernel(
    const float* __restrict__ A, const float* __restrict__ W,
    const float* __restrict__ bias, int M, float scale, int mode,
    float* __restrict__ Cf, __bf16* __restrict__ Cb) {
  __shared__ alignas(16) __bf16 As[2][64][32];
  __shared__ alignas(16) __bf16 Ws[2][64][32];
  const int row0 = blockIdx.x * 64;
  const int col0 = blockIdx.y * 64;
  const int tid  = threadIdx.x;
  const int wave = tid >> 5, lane = tid & 31;
  const int ti   = wave >> 1;          // M tile 0..3
  const int tj0  = (wave & 1) * 2;     // N tile pair base (0 or 2)
  const int r15  = lane & 15, half = lane >> 4;
  // staging coords: each thread owns an 8-wide chunk of one row
  const int sr = tid >> 2;             // 0..63
  const int sc = (tid & 3) * 8;        // 0,8,16,24
  int ga = row0 + sr;
  if (ga >= M) ga = M - 1;             // clamp; out-of-range rows never stored
  const float* ap = A + (size_t)ga * 256 + sc;
  const float* wp = W + (size_t)(col0 + sr) * 256 + sc;
  v8f acc0 = {};
  v8f acc1 = {};
  #pragma unroll 2
  for (int kk = 0; kk < 8; ++kk) {
    const int b = kk & 1;
    v4f a0 = *(const v4f*)(ap + kk * 32);
    v4f a1 = *(const v4f*)(ap + kk * 32 + 4);
    v4f w0 = *(const v4f*)(wp + kk * 32);
    v4f w1 = *(const v4f*)(wp + kk * 32 + 4);
    v8bf pa, pw;
    #pragma unroll
    for (int i = 0; i < 4; ++i) {
      pa[i] = (__bf16)a0[i]; pa[i + 4] = (__bf16)a1[i];
      pw[i] = (__bf16)w0[i]; pw[i + 4] = (__bf16)w1[i];
    }
    *(v8bf*)&As[b][sr][sc] = pa;
    *(v8bf*)&Ws[b][sr][sc] = pw;
    __syncthreads();
    v16bf af  = load2x8(&As[b][ti * 16 + r15][half * 16],
                        &As[b][ti * 16 + r15][half * 16 + 8]);
    v16bf bf0 = load2x8(&Ws[b][tj0 * 16 + r15][half * 16],
                        &Ws[b][tj0 * 16 + r15][half * 16 + 8]);
    v16bf bf1 = load2x8(&Ws[b][(tj0 + 1) * 16 + r15][half * 16],
                        &Ws[b][(tj0 + 1) * 16 + r15][half * 16 + 8]);
    acc0 = wmma_bf16(af, bf0, acc0);
    acc1 = wmma_bf16(af, bf1, acc1);
  }
  const int c0 = col0 + tj0 * 16 + r15;
  const int c1 = c0 + 16;
  const float b0v = bias[c0], b1v = bias[c1];
  #pragma unroll
  for (int rr = 0; rr < 8; ++rr) {
    int row = row0 + ti * 16 + half * 8 + rr;
    if (row < M) {
      float o0 = (acc0[rr] + b0v) * scale;
      float o1 = (acc1[rr] + b1v) * scale;
      if (mode == 0) {
        Cf[(size_t)row * 256 + c0] = o0;
        Cf[(size_t)row * 256 + c1] = o1;
      } else if (mode == 1) {
        Cb[(size_t)row * 256 + c0] = (__bf16)o0;
        Cb[(size_t)row * 256 + c1] = (__bf16)o1;
      } else {
        // VT layout: row = n*9216 + h*96 + w ; col = e*32 + d
        int n2 = row / 9216;
        int hw = row - n2 * 9216;
        int h  = hw / 96;
        int w  = hw - h * 96;
        size_t base = (((size_t)n2 * 96 + h) * 8);
        Cb[((base + (c0 >> 5)) * 32 + (c0 & 31)) * 96 + w] = (__bf16)o0;
        Cb[((base + (c1 >> 5)) * 32 + (c1 & 31)) * 96 + w] = (__bf16)o1;
      }
    }
  }
}

// ---------------------------------------------------------------------------
// Kernel 3: logits (K=32 -> one WMMA per tile) + softmax over 96.
// grid (ltile=5, e=8, n=8). Q: (n*300+l,256) bf16, Kp: (n*96+x,256) bf16.
// Aout: [((n*8+e)*300+l)*96 + x] bf16 (softmaxed probs).
// ---------------------------------------------------------------------------
__global__ __launch_bounds__(256) void attn_softmax_kernel(
    const __bf16* __restrict__ Q, const __bf16* __restrict__ Kp,
    __bf16* __restrict__ Aout) {
  __shared__ float ls[64][96];
  const int lt = blockIdx.x, e = blockIdx.y, n = blockIdx.z;
  const int lbase = lt * 64;
  const int tid = threadIdx.x;
  const int wave = tid >> 5, lane = tid & 31;
  const int ti = wave >> 1;
  const int r15 = lane & 15, half = lane >> 4;
  int lrow = lbase + ti * 16 + r15;
  if (lrow > 299) lrow = 299;                       // clamp reads; skip on store
  const __bf16* qp = Q + (size_t)(n * 300 + lrow) * 256 + e * 32;
  v16bf af = load2x8(qp + half * 16, qp + half * 16 + 8);
  #pragma unroll
  for (int jj = 0; jj < 3; ++jj) {
    int tj = (wave & 1) * 3 + jj;                   // 6 N tiles across wave parity
    int xcol = tj * 16 + r15;
    const __bf16* kp = Kp + (size_t)(n * 96 + xcol) * 256 + e * 32 + half * 16;
    v16bf bf = load2x8(kp, kp + 8);
    v8f acc = {};
    acc = wmma_bf16(af, bf, acc);
    #pragma unroll
    for (int rr = 0; rr < 8; ++rr)
      ls[ti * 16 + half * 8 + rr][tj * 16 + r15] = acc[rr];
  }
  __syncthreads();
  if (tid < 64) {
    int l = lbase + tid;
    if (l < 300) {
      float m = -3.4e38f;
      for (int w = 0; w < 96; ++w) m = fmaxf(m, ls[tid][w]);
      float s = 0.f;
      for (int w = 0; w < 96; ++w) {
        float ev = __expf(ls[tid][w] - m);
        ls[tid][w] = ev;
        s += ev;
      }
      float inv = 1.f / s;
      __bf16* dst = Aout + ((size_t)(n * 8 + e) * 300 + l) * 96;
      for (int w = 0; w < 96; ++w) dst[w] = (__bf16)(ls[tid][w] * inv);
    }
  }
}

// ---------------------------------------------------------------------------
// Kernel 4: fused attention apply.
// out[n,e,l,d] = sum_{h,w} a_col[l,h]*a_row[l,w]*v[n,h,w,e*32+d]
// = GEMM M=300(l) N=32(d) K=9216(h*96+w). A built on the fly into
// double-buffered LDS; B fragments read straight from global VT[n][h][e][d][w]
// (two contiguous 16B loads per lane). One barrier per K-step, 288 WMMAs/wave.
// ---------------------------------------------------------------------------
__global__ __launch_bounds__(256) void apply_kernel(
    const __bf16* __restrict__ Arow, const __bf16* __restrict__ Acol,
    const __bf16* __restrict__ VT, float* __restrict__ X2) {
  __shared__ alignas(16) __bf16 As[2][64][32];
  const int lt = blockIdx.x, e = blockIdx.y, n = blockIdx.z;
  const int lbase = lt * 64;
  const int tid = threadIdx.x;
  const int wave = tid >> 5, lane = tid & 31;
  const int ti = wave >> 1, tj = wave & 1;
  const int r15 = lane & 15, half = lane >> 4;
  // staging coords
  const int sr = tid >> 2;             // row 0..63
  const int sc = (tid & 3) * 8;        // 0,8,16,24
  int lclamp = lbase + sr;
  if (lclamp > 299) lclamp = 299;
  const __bf16* arp = Arow + ((size_t)(n * 8 + e) * 300 + lclamp) * 96;
  const __bf16* acp = Acol + ((size_t)(n * 8 + e) * 300 + lclamp) * 96;
  // B fragment base: lane owns column d, 16 consecutive w
  const int dl = tj * 16 + r15;
  const __bf16* vtb = VT + (((size_t)n * 96 * 8 + e) * 32 + dl) * 96 + half * 16;
  // h stride in VT elements: 8*32*96
  v8f acc = {};
  for (int kb = 0; kb < 288; ++kb) {
    const int h  = kb / 3;
    const int w0 = (kb - h * 3) * 32;
    const int b  = kb & 1;
    // A stage: pa[i] = a_col[l,h] * a_row[l,w0+sc+i]
    float acv = (float)acp[h];
    v8bf av = *(const v8bf*)(arp + w0 + sc);
    v8bf pa;
    #pragma unroll
    for (int i = 0; i < 8; ++i) pa[i] = (__bf16)(acv * (float)av[i]);
    *(v8bf*)&As[b][sr][sc] = pa;
    __syncthreads();
    const __bf16* bp = vtb + (size_t)h * (8 * 32 * 96) + w0;
    v16bf bf = load2x8(bp, bp + 8);
    v16bf af = load2x8(&As[b][ti * 16 + r15][half * 16],
                       &As[b][ti * 16 + r15][half * 16 + 8]);
    acc = wmma_bf16(af, bf, acc);
  }
  #pragma unroll
  for (int rr = 0; rr < 8; ++rr) {
    int l = lbase + ti * 16 + half * 8 + rr;
    if (l < 300) {
      int d = tj * 16 + r15;
      X2[((size_t)l * 8 + n) * 256 + e * 32 + d] = acc[rr];
    }
  }
}

// ---------------------------------------------------------------------------
extern "C" void kernel_launch(void* const* d_in, const int* in_sizes, int n_in,
                              void* d_out, int out_size, void* d_ws, size_t ws_size,
                              hipStream_t stream) {
  (void)in_sizes; (void)n_in; (void)out_size; (void)ws_size;
  const float* query_row = (const float*)d_in[0];
  const float* query_col = (const float*)d_in[1];
  const float* key_row   = (const float*)d_in[2];
  const float* key_col   = (const float*)d_in[3];
  const float* value     = (const float*)d_in[4];
  const float* Wall      = (const float*)d_in[5];   // (1280,256)
  const float* ball      = (const float*)d_in[6];   // (1280,)
  const float* Wout      = (const float*)d_in[7];   // (256,256)
  const float* bout      = (const float*)d_in[8];   // (256,)

  char* ws = (char*)d_ws;
  size_t off = 0;
  auto alloc = [&](size_t bytes) -> char* {
    char* p = ws + off;
    off += (bytes + 255) & ~(size_t)255;
    return p;
  };
  float*  km_row = (float*)alloc((size_t)8 * 96 * 256 * 4);
  float*  km_col = (float*)alloc((size_t)8 * 96 * 256 * 4);
  __bf16* qr     = (__bf16*)alloc((size_t)2400 * 256 * 2);
  __bf16* qc     = (__bf16*)alloc((size_t)2400 * 256 * 2);
  __bf16* krp    = (__bf16*)alloc((size_t)768 * 256 * 2);
  __bf16* kcp    = (__bf16*)alloc((size_t)768 * 256 * 2);
  __bf16* vt     = (__bf16*)alloc((size_t)8 * 96 * 8 * 32 * 96 * 2); // [n][h][e][d][w]
  __bf16* arow   = (__bf16*)alloc((size_t)64 * 300 * 96 * 2);
  __bf16* acol   = (__bf16*)alloc((size_t)64 * 300 * 96 * 2);
  float*  X2     = (float*)alloc((size_t)2400 * 256 * 4);

  // 1) key means (projection commutes with mean)
  means_kernel<<<768, 256, 0, stream>>>(key_row, key_col, km_row, km_col);

  const float scale = 0.17677669529663687f;  // 32^-0.5
  // 2) projections (bf16 outputs feed WMMA stages)
  gemm_proj_kernel<<<dim3(38, 4), 256, 0, stream>>>(query_row, Wall + 0 * 65536, ball + 0,    2400,  scale, 1, nullptr, qr);
  gemm_proj_kernel<<<dim3(38, 4), 256, 0, stream>>>(query_col, Wall + 1 * 65536, ball + 256,  2400,  scale, 1, nullptr, qc);
  gemm_proj_kernel<<<dim3(12, 4), 256, 0, stream>>>(km_row,    Wall + 2 * 65536, ball + 512,   768,  1.f,   1, nullptr, krp);
  gemm_proj_kernel<<<dim3(12, 4), 256, 0, stream>>>(km_col,    Wall + 3 * 65536, ball + 768,   768,  1.f,   1, nullptr, kcp);
  gemm_proj_kernel<<<dim3(1152, 4), 256, 0, stream>>>(value,   Wall + 4 * 65536, ball + 1024, 73728, 1.f,   2, nullptr, vt);

  // 3) row/col attention logits + softmax
  attn_softmax_kernel<<<dim3(5, 8, 8), 256, 0, stream>>>(qr, krp, arow);
  attn_softmax_kernel<<<dim3(5, 8, 8), 256, 0, stream>>>(qc, kcp, acol);

  // 4) fused separable-attention apply (K=9216 GEMM per (n,head))
  apply_kernel<<<dim3(5, 8, 8), 256, 0, stream>>>(arow, acol, vt, X2);

  // 5) output projection -> d_out (L,N,E) f32
  gemm_proj_kernel<<<dim3(38, 4), 256, 0, stream>>>(X2, Wout, bout, 2400, 1.f, 0,
                                                    (float*)d_out, nullptr);
}
